// Deform_20349555048985
// MI455X (gfx1250) — matrix operand order repaired and checked by reading, not compiled, not run on hardware
//
#include <hip/hip_runtime.h>
#include <hip/hip_bf16.h>
#include <math.h>

// ---------------------------------------------------------------------------
// Deformable conv v2, fully fused for gfx1250 (MI455X, wave32, WMMA).
//   B=8, C=64, H=W=128, OUT=128, K=9 taps, PAD=1
// One workgroup (256 thr = 8 wave32) handles one (b, y) row of 128 pixels.
// Phase 1: offset/mask 3x3 conv as WMMA GEMM (N=32, K=576), results in LDS.
// Phase 2: per tap, bilinear gather * mask -> f16 LDS tile (double-buffered),
//          WMMA GEMM (M=128, N=128, K=576) accumulated in registers, fp32 out.
// ---------------------------------------------------------------------------

typedef __attribute__((ext_vector_type(16))) _Float16 v16h;
typedef __attribute__((ext_vector_type(8)))  _Float16 v8h;
typedef __attribute__((ext_vector_type(8)))  float    v8f;

#define BB   8
#define CC   64
#define HH   128
#define WW   128
#define OUTC 128
#define KT   9          // 3x3 taps
#define NPAD 32         // 18 offset + 9 mask channels, padded to 32
#define APAD (CC + 8)   // LDS row stride in halves (bank spread, 16B aligned)

// workspace layout (halves), 32B-aligned sub-buffers
#define WCONV_HALVES (KT * 2 * OUTC * 32)   // [tap][s][n][32]  73728
#define WOM_OFF_B    (WCONV_HALVES * 2)     // byte offset of wom16 = 147456

// ---------------------------------------------------------------------------
// Prep: swizzle weights into per-lane B-fragment order, fp32 -> f16.
// B frag layout: halves[(tap*2+s)*N + n][32] ; lane reads 16 halves at +16*h
// giving K = 32s + 16h + 0..15 (contiguous order, same as the A fragments).
// ---------------------------------------------------------------------------
__global__ void dcn_prep_weights(const float* __restrict__ w_offset,
                                 const float* __restrict__ w_mask,
                                 const float* __restrict__ w_conv,
                                 _Float16* __restrict__ wconv16,
                                 _Float16* __restrict__ wom16) {
  int idx = blockIdx.x * blockDim.x + threadIdx.x;

  if (idx < KT * CC * OUTC) {                 // 73728 main-conv weights
    int n   = idx & (OUTC - 1);
    int c   = (idx >> 7) & (CC - 1);
    int tap = idx >> 13;                      // idx / (128*64)
    int s   = c >> 5;
    float v = w_conv[(size_t)(n * CC + c) * KT + tap];
    wconv16[((tap * 2 + s) * OUTC + n) * 32 + (c & 31)] = (_Float16)v;
  }

  if (idx < KT * CC * NPAD) {                 // 18432 offset/mask weights
    int n   = idx & (NPAD - 1);
    int c   = (idx >> 5) & (CC - 1);
    int tap = idx >> 11;                      // idx / (32*64)
    int s   = c >> 5;
    float v = 0.0f;
    if (n < 18)      v = w_offset[(size_t)(n * CC + c) * KT + tap];
    else if (n < 27) v = w_mask[(size_t)((n - 18) * CC + c) * KT + tap];
    wom16[((tap * 2 + s) * NPAD + n) * 32 + (c & 31)] = (_Float16)v;
  }
}

// Load an A fragment (16x32 f16) for m-row `row` in the LDS tile, k-step s.
// Lane group h (lane>>4): halves K = 32s+16h..+15 (contiguous, matches B).
__device__ __forceinline__ v16h load_afrag(const _Float16* __restrict__ row,
                                           int s, int h) {
  const v8h lo = *(const v8h*)(row + 32 * s + 16 * h);
  const v8h hi = *(const v8h*)(row + 32 * s + 16 * h + 8);
  v16h a;
#pragma unroll
  for (int i = 0; i < 8; ++i) { a[i] = lo[i]; a[i + 8] = hi[i]; }
  return a;
}

__global__ __launch_bounds__(256)
void dcn_fused_kernel(const float* __restrict__ x,
                      const float* __restrict__ b_offset,
                      const float* __restrict__ b_mask,
                      const _Float16* __restrict__ wconv16,
                      const _Float16* __restrict__ wom16,
                      float* __restrict__ out) {
  __shared__ _Float16 Atile[2][WW][APAD];    // double-buffered im2col tile
  __shared__ float offs_lds[18][WW];         // dy/dx per tap (ch 2k, 2k+1)
  __shared__ float mask_lds[KT][WW];         // sigmoid mask per tap

  const int tid  = threadIdx.x;
  const int wave = tid >> 5;                 // 0..7 (wave32!)
  const int lane = tid & 31;
  const int lm   = lane & 15;                // fragment column / row-in-tile
  const int h    = lane >> 4;                // lane half-group

  const int y = blockIdx.x & (HH - 1);
  const int b = blockIdx.x >> 7;             // blockIdx.x / HH

  const int p   = tid & (WW - 1);            // pixel owned for LDS fills
  const int grp = tid >> 7;                  // channel half: 0 -> c<32, 1 -> c>=32
  const float* xb = x + (size_t)b * CC * HH * WW;

  // ---------------- Phase 1: offset/mask conv via WMMA -------------------
  v8f acc0 = {};                             // channels 0..15
  v8f acc1 = {};                             // channels 16..31
#pragma unroll 1
  for (int k = 0; k < KT; ++k) {
    const int ky = k / 3, kx = k % 3;
    _Float16 (* __restrict__ At)[APAD] = Atile[k & 1];
    // identity im2col fill: A[p][c] = x[b,c,y+ky-1,p+kx-1] (0 if OOB).
    // Branch-free: clamp address, scale by validity; 8-wide packed DS stores.
    {
      const int yy = y + ky - 1;
      const int xx = p + kx - 1;
      const float okf =
          ((unsigned)yy < (unsigned)HH && (unsigned)xx < (unsigned)WW) ? 1.0f : 0.0f;
      const int cyy = min(max(yy, 0), HH - 1);
      const int cxx = min(max(xx, 0), WW - 1);
      const float* src = xb + ((size_t)(grp * 32) * HH + cyy) * WW + cxx;
#pragma unroll
      for (int c0 = 0; c0 < 32; c0 += 8) {
        v8h pk;
#pragma unroll
        for (int i = 0; i < 8; ++i)
          pk[i] = (_Float16)(okf * src[(size_t)(c0 + i) * HH * WW]);
        *(v8h*)&At[p][grp * 32 + c0] = pk;
      }
    }
    __syncthreads();                          // fill(k) done; reads of buf(k-2) done

    const _Float16* arow = &At[wave * 16 + lm][0];
#pragma unroll
    for (int s = 0; s < 2; ++s) {
      v16h a  = load_afrag(arow, s, h);
      v16h b0 = *(const v16h*)(wom16 + ((k * 2 + s) * NPAD + lm) * 32 + 16 * h);
      v16h b1 = *(const v16h*)(wom16 + ((k * 2 + s) * NPAD + 16 + lm) * 32 + 16 * h);
      acc0 = __builtin_amdgcn_wmma_f32_16x16x32_f16(false, a, false, b0,
                                                    (short)0, acc0, false, false);
      acc1 = __builtin_amdgcn_wmma_f32_16x16x32_f16(false, a, false, b1,
                                                    (short)0, acc1, false, false);
    }
  }
  __syncthreads();

  // epilogue: bias, sigmoid(mask), park in LDS. C/D map: m = r + 8h, n = lm.
  {
    const int pix_base = wave * 16 + 8 * h;
#pragma unroll
    for (int r = 0; r < 8; ++r) {
      const int pp = pix_base + r;
      // n-tile 0: channels 0..15 -> always offset channels
      offs_lds[lm][pp] = acc0[r] + b_offset[lm];
      // n-tile 1: channels 16..31
      const int ch1 = 16 + lm;
      if (ch1 < 18) {
        offs_lds[ch1][pp] = acc1[r] + b_offset[ch1];
      } else if (ch1 < 27) {
        float v = acc1[r] + b_mask[ch1 - 18];
        mask_lds[ch1 - 18][pp] = 1.0f / (1.0f + __expf(-v));
      }
    }
  }
  __syncthreads();

  // ---------------- Phase 2: deformable gather + main GEMM ---------------
  v8f acc[8] = {};                           // 8 M-subtiles x (16x16 f32)
#pragma unroll 1
  for (int k = 0; k < KT; ++k) {
    const int ky = k / 3, kx = k % 3;
    _Float16 (* __restrict__ At)[APAD] = Atile[k & 1];
    // bilinear-sample fill: per-pixel corner math once, 32 channels each,
    // branch-free (validity folded into weights), packed 16B DS stores.
    {
      const float dy  = offs_lds[2 * k][p];
      const float dx  = offs_lds[2 * k + 1][p];
      const float msk = mask_lds[k][p];
      const float gy = (float)(y - 1 + ky) + dy;
      const float gx = (float)(p - 1 + kx) + dx;
      const float fy = floorf(gy), fx = floorf(gx);
      const float wy = gy - fy,    wx = gx - fx;
      const int iy0 = (int)fy, ix0 = (int)fx;
      const int iy1 = iy0 + 1, ix1 = ix0 + 1;
      const float vy0 = ((unsigned)iy0 < (unsigned)HH) ? 1.0f : 0.0f;
      const float vy1 = ((unsigned)iy1 < (unsigned)HH) ? 1.0f : 0.0f;
      const float vx0 = ((unsigned)ix0 < (unsigned)WW) ? 1.0f : 0.0f;
      const float vx1 = ((unsigned)ix1 < (unsigned)WW) ? 1.0f : 0.0f;
      const float w00 = (1.f - wy) * (1.f - wx) * vy0 * vx0 * msk;
      const float w01 = (1.f - wy) * wx         * vy0 * vx1 * msk;
      const float w10 = wy * (1.f - wx)         * vy1 * vx0 * msk;
      const float w11 = wy * wx                 * vy1 * vx1 * msk;
      const int cy0 = min(max(iy0, 0), HH - 1), cy1 = min(max(iy1, 0), HH - 1);
      const int cx0 = min(max(ix0, 0), WW - 1), cx1 = min(max(ix1, 0), WW - 1);
      const int o00 = cy0 * WW + cx0, o01 = cy0 * WW + cx1;
      const int o10 = cy1 * WW + cx0, o11 = cy1 * WW + cx1;
      const float* src = xb + (size_t)(grp * 32) * HH * WW;
#pragma unroll
      for (int c0 = 0; c0 < 32; c0 += 8) {
        v8h pk;
#pragma unroll
        for (int i = 0; i < 8; ++i) {
          const float* xc = src + (size_t)(c0 + i) * HH * WW;
          pk[i] = (_Float16)(w00 * xc[o00] + w01 * xc[o01] +
                             w10 * xc[o10] + w11 * xc[o11]);
        }
        *(v8h*)&At[p][grp * 32 + c0] = pk;
      }
    }
    // prefetch next tap's B fragments into cache (global_prefetch_b8)
    if (k + 1 < KT)
      __builtin_prefetch(wconv16 + ((k + 1) * 2 * OUTC + wave * 16 + lm) * 32, 0, 3);
    __syncthreads();                          // fill(k) done; reads of buf(k-2) done

#pragma unroll
    for (int s = 0; s < 2; ++s) {
      const v16h bf = *(const v16h*)(wconv16 +
          ((k * 2 + s) * OUTC + wave * 16 + lm) * 32 + 16 * h);
#pragma unroll
      for (int mt = 0; mt < 8; ++mt) {
        v16h a = load_afrag(&At[mt * 16 + lm][0], s, h);
        acc[mt] = __builtin_amdgcn_wmma_f32_16x16x32_f16(false, a, false, bf,
                                                         (short)0, acc[mt],
                                                         false, false);
      }
    }
  }

  // ---------------- write 128x128 fp32 tile ------------------------------
  const int n = wave * 16 + lm;              // output channel
  float* ob = out + ((size_t)(b * OUTC + n) * HH + y) * WW;
#pragma unroll
  for (int mt = 0; mt < 8; ++mt) {
#pragma unroll
    for (int r = 0; r < 8; ++r) {
      ob[mt * 16 + 8 * h + r] = acc[mt][r];
    }
  }
}

extern "C" void kernel_launch(void* const* d_in, const int* in_sizes, int n_in,
                              void* d_out, int out_size, void* d_ws, size_t ws_size,
                              hipStream_t stream) {
  const float* x        = (const float*)d_in[0];
  const float* w_offset = (const float*)d_in[1];
  const float* b_offset = (const float*)d_in[2];
  const float* w_mask   = (const float*)d_in[3];
  const float* b_mask   = (const float*)d_in[4];
  const float* w_conv   = (const float*)d_in[5];
  float* out = (float*)d_out;

  _Float16* wconv16 = (_Float16*)d_ws;                       // 147456 B
  _Float16* wom16   = (_Float16*)((char*)d_ws + WOM_OFF_B);  //  36864 B

  // swizzle weights to f16 fragment order
  dcn_prep_weights<<<(KT * CC * OUTC + 255) / 256, 256, 0, stream>>>(
      w_offset, w_mask, w_conv, wconv16, wom16);

  // fused deformable conv: one WG per (b, y) row
  dcn_fused_kernel<<<BB * HH, 256, 0, stream>>>(
      x, b_offset, b_mask, wconv16, wom16, out);
}
